// HighResolutionHeadWithCrissCrossAttention_90804198572471
// MI455X (gfx1250) — matrix operand order repaired and checked by reading, not compiled, not run on hardware
//
#include <hip/hip_runtime.h>
#include <hip/hip_bf16.h>

typedef __attribute__((ext_vector_type(16))) _Float16 v16h;
typedef __attribute__((ext_vector_type(8)))  float    v8f;
typedef __attribute__((ext_vector_type(4)))  float    f4;

#define NBATCH 2
#define LOG2E 1.4426950408889634f

// ---------------------------------------------------------------------------
// Kernel 1: per-scale prep.
//   y  = conv1x1(x, w, b)            [B,8,N] f32
//   q  = conv1x1(y, wq, bq)          [B,N]   f32   (scalar per pixel, d=1)
//   k  = conv1x1(y, wk, bk)          [B,N]   f32
//   vT [B,16,N32] f16: rows 0..7 = v, row 8 = ones (n<N), rest/pad = 0.
// Zero pad columns + zero rows 9..15 make the WMMA chunk loop branchless.
// ---------------------------------------------------------------------------
__global__ void prep_kernel(const float* __restrict__ x, int Cin, int N, int N32,
                            const float* __restrict__ w,  const float* __restrict__ bias,
                            const float* __restrict__ wq, const float* __restrict__ bq,
                            const float* __restrict__ wk, const float* __restrict__ bk,
                            const float* __restrict__ wv, const float* __restrict__ bv,
                            float* __restrict__ yp, float* __restrict__ qArr,
                            float* __restrict__ kArr, _Float16* __restrict__ vT) {
    int idx = blockIdx.x * blockDim.x + threadIdx.x;
    if (idx >= NBATCH * N32) return;
    int b = idx / N32;
    int n = idx - b * N32;

    if (n >= N) {                       // pad column: all 16 rows zero
#pragma unroll
        for (int c = 0; c < 16; ++c)
            vT[((size_t)(b * 16 + c)) * N32 + n] = (_Float16)0.0f;
        return;
    }

    float y[8];
#pragma unroll
    for (int c = 0; c < 8; ++c) y[c] = bias[c];
    for (int ci = 0; ci < Cin; ++ci) {
        float xv = x[((size_t)(b * Cin + ci)) * N + n];
#pragma unroll
        for (int c = 0; c < 8; ++c) y[c] += w[c * Cin + ci] * xv;
    }

    float qv = bq[0], kv = bk[0];
#pragma unroll
    for (int c = 0; c < 8; ++c) { qv += wq[c] * y[c]; kv += wk[c] * y[c]; }

#pragma unroll
    for (int c = 0; c < 8; ++c) {
        float vv = bv[c];
#pragma unroll
        for (int cc = 0; cc < 8; ++cc) vv += wv[c * 8 + cc] * y[cc];
        vT[((size_t)(b * 16 + c)) * N32 + n] = (_Float16)vv;
        yp[((size_t)(b * 8 + c)) * N + n] = y[c];
    }
    vT[((size_t)(b * 16 + 8)) * N32 + n] = (_Float16)1.0f;  // denominator column
#pragma unroll
    for (int c = 9; c < 16; ++c)
        vT[((size_t)(b * 16 + c)) * N32 + n] = (_Float16)0.0f;
    qArr[(size_t)b * N + n] = qv;
    kArr[(size_t)b * N + n] = kv;
}

// ---------------------------------------------------------------------------
// Kernel 2: per-batch k min/max (softmax stabilization: logits = s*k).
// ---------------------------------------------------------------------------
__global__ void kminmax_kernel(const float* __restrict__ kArr, int N,
                               float* __restrict__ kmm) {
    __shared__ float smx[256], smn[256];
    int b = blockIdx.x;
    int tid = threadIdx.x;
    float mx = -3.4e38f, mn = 3.4e38f;
    for (int i = tid; i < N; i += 256) {
        float v = kArr[(size_t)b * N + i];
        mx = fmaxf(mx, v); mn = fminf(mn, v);
    }
    smx[tid] = mx; smn[tid] = mn;
    __syncthreads();
    for (int st = 128; st > 0; st >>= 1) {
        if (tid < st) {
            smx[tid] = fmaxf(smx[tid], smx[tid + st]);
            smn[tid] = fminf(smn[tid], smn[tid + st]);
        }
        __syncthreads();
    }
    if (tid == 0) { kmm[b * 2 + 0] = smx[0]; kmm[b * 2 + 1] = smn[0]; }
}

// ---------------------------------------------------------------------------
// Kernel 3: attention + gamma conv + residual. Branchless WMMA chunk loop:
//   A[m,kk]  = exp2(s2_m * k[base+kk] - m2_m)    (16x32 f16, contiguous k = 16h+s)
//   B[kk,n]  = vT[n][base+kk]  (n=0..7 v, n=8 ones, n>8 / pad cols = 0)
//   D += A*B : D[m,0..7] = numerators, D[m,8] = denominator
//   yf[c,q]  = yp[c,q] + bg[c] + sum_c' wg[c,c'] * D[m,c']/D[m,8]
// Per A element: v_fma_f32 + raw v_exp_f32 (__builtin_amdgcn_exp2f); logits
// are <= 0 by construction so the fast exp2 path is range-safe.
// ---------------------------------------------------------------------------
__global__ void __launch_bounds__(256) attn_kernel(
        int N, int N32, int nQB,
        const float* __restrict__ qArr, const float* __restrict__ kArr,
        const _Float16* __restrict__ vT, const float* __restrict__ yp,
        const float* __restrict__ wg, const float* __restrict__ bg,
        const float* __restrict__ kmm, float* __restrict__ yf) {
    __shared__ float kl[9216];
    __shared__ float Dsh[8][16][17];

    int b  = blockIdx.x / nQB;
    int qb = blockIdx.x - b * nQB;
    int tid  = threadIdx.x;
    int lane = tid & 31;
    int wid  = tid >> 5;

    // Stage keys (zero-padded to N32) into LDS.
    for (int i = tid; i < N32; i += 256)
        kl[i] = (i < N) ? kArr[(size_t)b * N + i] : 0.0f;
    __syncthreads();

    int q0 = qb * 128 + wid * 16;
    if (q0 >= N) return;            // wave-uniform: EXEC all-ones below

    int m  = lane & 15;             // A row (query within tile)
    int qi = q0 + m;                // N multiple of 16 -> in range
    float s    = qArr[(size_t)b * N + qi];
    float kmax = kmm[b * 2 + 0], kmin = kmm[b * 2 + 1];
    float mst  = fmaxf(s * kmax, s * kmin);   // logits <= 0 -> exp in (0,1]
    float s2 = s * LOG2E;                     // fold log2(e): exp(x)=exp2(x*log2e)
    float m2 = mst * LOG2E;

    int hi8   = (lane >= 16) ? 8  : 0;   // unused; A also uses off16 (upper lanes hold K+16)
    int off16 = (lane >= 16) ? 16 : 0;   // B layout: upper lanes hold K+16
    int ncol  = lane & 15;               // B column (v channel / ones / zero)
    const _Float16* vrow = vT + ((size_t)(b * 16 + ncol)) * N32;
    const f4* kl4 = (const f4*)kl;

    v8f acc = {};
    int nCh = N32 >> 5;
    for (int ch = 0; ch < nCh; ++ch) {
        int base = ch << 5;

        // 16 key values for this lane: two 8-float runs (ds_load_b128 x4).
        int i0 = (base + off16) >> 2;      // 16B aligned
        int i1 = (base + 8 + off16) >> 2;
        f4 k0 = kl4[i0];
        f4 k1 = kl4[i0 + 1];
        f4 k2 = kl4[i1];
        f4 k3 = kl4[i1 + 1];
        float kv[16] = {k0[0], k0[1], k0[2], k0[3], k1[0], k1[1], k1[2], k1[3],
                        k2[0], k2[1], k2[2], k2[3], k3[0], k3[1], k3[2], k3[3]};

        v16h a;
#pragma unroll
        for (int i = 0; i < 16; ++i)
            a[i] = (_Float16)__builtin_amdgcn_exp2f(s2 * kv[i] - m2); // fma+exp

        // B fragment: 16 contiguous halves (32B aligned) -> 2x b128 loads.
        v16h bb = *(const v16h*)(vrow + base + off16);

        acc = __builtin_amdgcn_wmma_f32_16x16x32_f16(
                  false, a, false, bb, (short)0, acc, false, false);
    }

    // Spill D to LDS (wave-private slice; same-wave RAW via dscnt).
#pragma unroll
    for (int r = 0; r < 8; ++r) {
        int row = (lane < 16) ? r : r + 8;
        Dsh[wid][row][ncol] = acc[r];
    }

    // Finish: softmax divide, gamma conv, residual. 16 queries x 8 channels.
#pragma unroll
    for (int t = 0; t < 4; ++t) {
        int it = lane + 32 * t;
        int mm = it >> 3;
        int co = it & 7;
        int qq = q0 + mm;
        float inv = 1.0f / Dsh[wid][mm][8];
        float r = bg[co];
#pragma unroll
        for (int c = 0; c < 8; ++c) r += wg[co * 8 + c] * (Dsh[wid][mm][c] * inv);
        size_t o = ((size_t)(b * 8 + co)) * N + qq;
        yf[o] = yp[o] + r;
    }
}

// ---------------------------------------------------------------------------
// Kernel 4: fused head. bilinear gather -> 32x32 conv -> BN -> ReLU -> 19x32.
// ---------------------------------------------------------------------------
__device__ __forceinline__ float bil(const float* __restrict__ yf, int b, int c,
                                     int Hs, int h, int w) {
    float scale = (float)Hs * (1.0f / 96.0f);
    float sy = ((float)h + 0.5f) * scale - 0.5f;
    float sx = ((float)w + 0.5f) * scale - 0.5f;
    sy = fminf(fmaxf(sy, 0.0f), (float)(Hs - 1));
    sx = fminf(fmaxf(sx, 0.0f), (float)(Hs - 1));
    int y0 = (int)sy, x0 = (int)sx;
    int y1 = min(y0 + 1, Hs - 1), x1 = min(x0 + 1, Hs - 1);
    float fy = sy - (float)y0, fx = sx - (float)x0;
    const float* p = yf + ((size_t)(b * 8 + c)) * Hs * Hs;
    float v00 = p[y0 * Hs + x0], v01 = p[y0 * Hs + x1];
    float v10 = p[y1 * Hs + x0], v11 = p[y1 * Hs + x1];
    return v00 * (1.0f - fy) * (1.0f - fx) + v01 * (1.0f - fy) * fx +
           v10 * fy * (1.0f - fx)          + v11 * fy * fx;
}

__global__ void head_kernel(const float* __restrict__ yf0, const float* __restrict__ yf1,
                            const float* __restrict__ yf2, const float* __restrict__ yf3,
                            const float* __restrict__ wl0, const float* __restrict__ bl0,
                            const float* __restrict__ bns, const float* __restrict__ bnb,
                            const float* __restrict__ bnm, const float* __restrict__ bnv,
                            const float* __restrict__ wl1, const float* __restrict__ bl1,
                            float* __restrict__ out) {
    int idx = blockIdx.x * blockDim.x + threadIdx.x;
    if (idx >= NBATCH * 9216) return;
    int b  = idx / 9216;
    int hw = idx - b * 9216;
    int h  = hw / 96, w = hw - h * 96;

    float cat[32];
#pragma unroll
    for (int c = 0; c < 8; ++c) {
        cat[c]      = yf0[((size_t)(b * 8 + c)) * 9216 + hw];
        cat[8 + c]  = bil(yf1, b, c, 48, h, w);
        cat[16 + c] = bil(yf2, b, c, 24, h, w);
        cat[24 + c] = bil(yf3, b, c, 12, h, w);
    }

    float tv[32];
    for (int o = 0; o < 32; ++o) {
        float acc = bl0[o];
#pragma unroll 8
        for (int c = 0; c < 32; ++c) acc += wl0[o * 32 + c] * cat[c];
        float sc = bns[o] * __frsqrt_rn(bnv[o] + 1e-5f);
        acc = (acc - bnm[o]) * sc + bnb[o];
        tv[o] = fmaxf(acc, 0.0f);
    }
    for (int o = 0; o < 19; ++o) {
        float acc = bl1[o];
#pragma unroll 8
        for (int c = 0; c < 32; ++c) acc += wl1[o * 32 + c] * tv[c];
        out[((size_t)(b * 19 + o)) * 9216 + hw] = acc;
    }
}

// ---------------------------------------------------------------------------
extern "C" void kernel_launch(void* const* d_in, const int* in_sizes, int n_in,
                              void* d_out, int out_size, void* d_ws, size_t ws_size,
                              hipStream_t stream) {
    (void)in_sizes; (void)n_in; (void)out_size; (void)ws_size;

    static const int Ns[4]   = {9216, 2304, 576, 144};
    static const int Cin[4]  = {48, 96, 192, 384};
    static const int ccaB[4] = {12, 20, 28, 20};   // scale 3 reuses cca1 (faithful)

    const float* x[4]  = {(const float*)d_in[0], (const float*)d_in[1],
                          (const float*)d_in[2], (const float*)d_in[3]};
    const float* w[4]  = {(const float*)d_in[4], (const float*)d_in[6],
                          (const float*)d_in[8], (const float*)d_in[10]};
    const float* bs[4] = {(const float*)d_in[5], (const float*)d_in[7],
                          (const float*)d_in[9], (const float*)d_in[11]};

    // Workspace bump allocator (256B aligned).
    char* ws = (char*)d_ws;
    size_t off = 0;
    auto alloc = [&](size_t bytes) -> char* {
        off = (off + 255) & ~(size_t)255;
        char* p = ws + off;
        off += bytes;
        return p;
    };

    float*    yp[4];  float* qA[4];  float* kA[4];
    _Float16* vT[4];  float* yf[4];  float* km[4];
    int N32s[4];
    for (int i = 0; i < 4; ++i) {
        size_t N = (size_t)Ns[i];
        size_t N32 = (N + 31) & ~(size_t)31;
        N32s[i] = (int)N32;
        yp[i] = (float*)   alloc(NBATCH * 8 * N * sizeof(float));
        qA[i] = (float*)   alloc(NBATCH * N * sizeof(float));
        kA[i] = (float*)   alloc(NBATCH * N * sizeof(float));
        vT[i] = (_Float16*)alloc(NBATCH * 16 * N32 * sizeof(_Float16));
        yf[i] = (float*)   alloc(NBATCH * 8 * N * sizeof(float));
        km[i] = (float*)   alloc(NBATCH * 2 * sizeof(float));
    }

    for (int i = 0; i < 4; ++i) {
        int N = Ns[i], N32 = N32s[i];
        int cb = ccaB[i];
        const float* wq = (const float*)d_in[cb + 0];
        const float* bq = (const float*)d_in[cb + 1];
        const float* wk = (const float*)d_in[cb + 2];
        const float* bk = (const float*)d_in[cb + 3];
        const float* wv = (const float*)d_in[cb + 4];
        const float* bv = (const float*)d_in[cb + 5];
        const float* wg = (const float*)d_in[cb + 6];
        const float* bg = (const float*)d_in[cb + 7];

        int total = NBATCH * N32;
        prep_kernel<<<(total + 255) / 256, 256, 0, stream>>>(
            x[i], Cin[i], N, N32, w[i], bs[i], wq, bq, wk, bk, wv, bv,
            yp[i], qA[i], kA[i], vT[i]);

        kminmax_kernel<<<NBATCH, 256, 0, stream>>>(kA[i], N, km[i]);

        int nQB = (N + 127) / 128;
        attn_kernel<<<NBATCH * nQB, 256, 0, stream>>>(
            N, N32, nQB, qA[i], kA[i], vT[i], yp[i], wg, bg, km[i], yf[i]);
    }

    head_kernel<<<(NBATCH * 9216 + 255) / 256, 256, 0, stream>>>(
        yf[0], yf[1], yf[2], yf[3],
        (const float*)d_in[36], (const float*)d_in[37],
        (const float*)d_in[38], (const float*)d_in[39],
        (const float*)d_in[40], (const float*)d_in[41],
        (const float*)d_in[42], (const float*)d_in[43],
        (float*)d_out);
}
